// BiDenseGeneral_6047313952874
// MI455X (gfx1250) — 8-bit matrix operand order repaired and checked by reading, compiled, not run on hardware
//
#include <hip/hip_runtime.h>
#include <hip/hip_bf16.h>


#define D_DIM 2048
#define F_DIM 8192
#define M_DIM 16384
#define FP32_EPS 1.1920928955078125e-07f

#define LDS_STRIDE 80
#define A_ROWS 128
#define B_ROWS 256
#define A_BYTES (A_ROWS * LDS_STRIDE)
#define B_BYTES (B_ROWS * LDS_STRIDE)
#define BUF_BYTES (A_BYTES + B_BYTES)

typedef int v8i __attribute__((ext_vector_type(8)));
typedef int v4i __attribute__((ext_vector_type(4)));
typedef int v2i __attribute__((ext_vector_type(2)));

__global__ __launch_bounds__(256) void bidense_pack_weights(
    const float* __restrict__ W, signed char* __restrict__ SwT,
    float* __restrict__ w_bound) {
  const int f = blockIdx.x * 256 + threadIdx.x;
  float m = 0.0f;
  signed char* dst = SwT + (size_t)f * D_DIM;
  for (int d = 0; d < D_DIM; d += 4) {
    float w0 = W[(size_t)(d + 0) * F_DIM + f];
    float w1 = W[(size_t)(d + 1) * F_DIM + f];
    float w2 = W[(size_t)(d + 2) * F_DIM + f];
    float w3 = W[(size_t)(d + 3) * F_DIM + f];
    m = fmaxf(m, fmaxf(fmaxf(fabsf(w0), fabsf(w1)), fmaxf(fabsf(w2), fabsf(w3))));
    int b0 = (w0 >= 0.0f) ? 0x01 : 0xFF;
    int b1 = (w1 >= 0.0f) ? 0x01 : 0xFF;
    int b2 = (w2 >= 0.0f) ? 0x01 : 0xFF;
    int b3 = (w3 >= 0.0f) ? 0x01 : 0xFF;
    *(int*)(dst + d) = b0 | (b1 << 8) | (b2 << 16) | (b3 << 24);
  }
  w_bound[f] = m + FP32_EPS;
}

__global__ __launch_bounds__(256) void bidense_pack_inputs(
    const float* __restrict__ x, signed char* __restrict__ Sx,
    float* __restrict__ a_bound) {
  const int lane = threadIdx.x & 31;
  const int wave = threadIdx.x >> 5;
  const int row = blockIdx.x * 8 + wave;
  const float4* xr = (const float4*)(x + (size_t)row * D_DIM + lane * 64);
  float m = 0.0f;
  int packed[16];
#pragma unroll
  for (int i = 0; i < 16; ++i) {
    float4 v = xr[i];
    m = fmaxf(m, fmaxf(fmaxf(fabsf(v.x), fabsf(v.y)), fmaxf(fabsf(v.z), fabsf(v.w))));
    int b0 = (v.x >= 0.0f) ? 0x01 : 0xFF;
    int b1 = (v.y >= 0.0f) ? 0x01 : 0xFF;
    int b2 = (v.z >= 0.0f) ? 0x01 : 0xFF;
    int b3 = (v.w >= 0.0f) ? 0x01 : 0xFF;
    packed[i] = b0 | (b1 << 8) | (b2 << 16) | (b3 << 24);
  }
#pragma unroll
  for (int off = 16; off > 0; off >>= 1) m = fmaxf(m, __shfl_xor(m, off, 32));
  if (lane == 0) a_bound[row] = m + FP32_EPS;
  int4* dst = (int4*)(Sx + (size_t)row * D_DIM + lane * 64);
#pragma unroll
  for (int i = 0; i < 4; ++i)
    dst[i] = make_int4(packed[4 * i + 0], packed[4 * i + 1],
                       packed[4 * i + 2], packed[4 * i + 3]);
}

__global__ __launch_bounds__(256) void bidense_wmma_gemm(
    const signed char* __restrict__ Sx, const signed char* __restrict__ SwT,
    const float* __restrict__ a_bound, const float* __restrict__ w_bound,
    const float* __restrict__ bias, float* __restrict__ out) {
  __shared__ signed char smem[2 * BUF_BYTES];

  const int lane = threadIdx.x & 31;
  const int wave = threadIdx.x >> 5;
  const int waveM = wave >> 2;
  const int waveN = wave & 3;
  const int mWG = blockIdx.y * 128;
  const int nWG = blockIdx.x * 256;
  const int m0 = mWG + waveM * 64;
  const int n0 = nWG + waveN * 64;

  unsigned aDst[2];
  unsigned long long aSrc[2];
#pragma unroll
  for (int q = 0; q < 2; ++q) {
    int c = (wave * 2 + q) * 32 + lane;
    int row = c >> 2, colb = (c & 3) * 16;
    aDst[q] = (unsigned)(row * LDS_STRIDE + colb);
    aSrc[q] = (unsigned long long)(Sx + (size_t)(mWG + row) * D_DIM + colb);
  }
  unsigned bDst[4];
  unsigned long long bSrc[4];
#pragma unroll
  for (int q = 0; q < 4; ++q) {
    int c = (wave * 4 + q) * 32 + lane;
    int row = c >> 2, colb = (c & 3) * 16;
    bDst[q] = (unsigned)(A_BYTES + row * LDS_STRIDE + colb);
    bSrc[q] = (unsigned long long)(SwT + (size_t)(nWG + row) * D_DIM + colb);
  }

  const unsigned smemBase = (unsigned)(uintptr_t)(void*)&smem[0];

  auto issue_tile = [&](int buf, int k) {
    const unsigned base = smemBase + (unsigned)(buf * BUF_BYTES);
#pragma unroll
    for (int q = 0; q < 2; ++q)
      asm volatile("global_load_async_to_lds_b128 %0, %1, off"
                   :: "v"(base + aDst[q]), "v"(aSrc[q] + (unsigned)k)
                   : "memory");
#pragma unroll
    for (int q = 0; q < 4; ++q)
      asm volatile("global_load_async_to_lds_b128 %0, %1, off"
                   :: "v"(base + bDst[q]), "v"(bSrc[q] + (unsigned)k)
                   : "memory");
  };

  const int r = lane & 15;
  const int aKb = (lane >> 4) * 16;
  const int bKg = (lane >> 4) * 16;
  unsigned aOff[4], bOff[4];
#pragma unroll
  for (int i = 0; i < 4; ++i)
    aOff[i] = (unsigned)((waveM * 64 + i * 16 + r) * LDS_STRIDE + aKb);
#pragma unroll
  for (int j = 0; j < 4; ++j)
    bOff[j] = (unsigned)(A_BYTES + (waveN * 64 + j * 16 + r) * LDS_STRIDE + bKg);

  v8i acc[4][4] = {};

  auto compute_tile = [&](const signed char* T) {
    v8i afr[4], bfr[4];
#pragma unroll
    for (int i = 0; i < 4; ++i) {
      v2i x0 = *(const v2i*)(T + aOff[i] + 0);
      v2i x1 = *(const v2i*)(T + aOff[i] + 8);
      v2i x2 = *(const v2i*)(T + aOff[i] + 32);
      v2i x3 = *(const v2i*)(T + aOff[i] + 40);
      afr[i][0] = x0[0]; afr[i][1] = x0[1];
      afr[i][2] = x1[0]; afr[i][3] = x1[1];
      afr[i][4] = x2[0]; afr[i][5] = x2[1];
      afr[i][6] = x3[0]; afr[i][7] = x3[1];
    }
#pragma unroll
    for (int j = 0; j < 4; ++j) {
      v4i y0 = *(const v4i*)(T + bOff[j] + 0);
      v4i y1 = *(const v4i*)(T + bOff[j] + 32);
      bfr[j][0] = y0[0]; bfr[j][1] = y0[1];
      bfr[j][2] = y0[2]; bfr[j][3] = y0[3];
      bfr[j][4] = y1[0]; bfr[j][5] = y1[1];
      bfr[j][6] = y1[2]; bfr[j][7] = y1[3];
    }
#pragma unroll
    for (int i = 0; i < 4; ++i)
#pragma unroll
      for (int j = 0; j < 4; ++j)
        acc[i][j] = __builtin_amdgcn_wmma_i32_16x16x64_iu8(
             true, afr[i],  true, bfr[j], acc[i][j],
             false,  false);
  };

  issue_tile(0, 0);
  issue_tile(1, 64);

  const signed char* T0 = smem;
  const signed char* T1 = smem + BUF_BYTES;

  for (int k0 = 0; k0 < D_DIM; k0 += 128) {
    asm volatile("s_wait_asynccnt 0x6" ::: "memory");
    __syncthreads();
    compute_tile(T0);
    __syncthreads();
    if (k0 + 128 < D_DIM) issue_tile(0, k0 + 128);

    if (k0 + 128 < D_DIM)
      asm volatile("s_wait_asynccnt 0x6" ::: "memory");
    else
      asm volatile("s_wait_asynccnt 0x0" ::: "memory");
    __syncthreads();
    compute_tile(T1);
    __syncthreads();
    if (k0 + 192 < D_DIM) issue_tile(1, k0 + 192);
  }

  const int colOff = lane & 15;
  const int rowHalf = (lane >> 4) * 8;
#pragma unroll
  for (int j = 0; j < 4; ++j) {
    const int ncol = n0 + j * 16 + colOff;
    const float wb = 0.25f * w_bound[ncol];
    const float bs = bias[ncol];
#pragma unroll
    for (int i = 0; i < 4; ++i) {
#pragma unroll
      for (int t = 0; t < 8; ++t) {
        const int mrow = m0 + i * 16 + rowHalf + t;
        out[(size_t)mrow * F_DIM + ncol] =
            a_bound[mrow] * wb * (float)acc[i][j][t] + bs;
      }
    }
  }
}

extern "C" void kernel_launch(void* const* d_in, const int* in_sizes, int n_in,
                              void* d_out, int out_size, void* d_ws,
                              size_t ws_size, hipStream_t stream) {
  const float* x = (const float*)d_in[0];
  const float* W = (const float*)d_in[1];
  const float* bias = (const float*)d_in[2];
  float* out = (float*)d_out;

  char* ws = (char*)d_ws;
  signed char* SwT = (signed char*)ws;
  signed char* Sx = (signed char*)(ws + (size_t)16 * 1024 * 1024);
  float* w_bound = (float*)(ws + (size_t)48 * 1024 * 1024);
  float* a_bound = (float*)(ws + (size_t)48 * 1024 * 1024 + 65536);

  bidense_pack_weights<<<F_DIM / 256, 256, 0, stream>>>(W, SwT, w_bound);
  bidense_pack_inputs<<<M_DIM / 8, 256, 0, stream>>>(x, Sx, a_bound);

  dim3 grid(F_DIM / 256, M_DIM / 128);
  bidense_wmma_gemm<<<grid, 256, 0, stream>>>(Sx, SwT, a_bound, w_bound, bias,
                                              out);
}
